// GNN_1_2_22385369546937
// MI455X (gfx1250) — matrix operand order repaired and checked by reading, not compiled, not run on hardware
//
#include <hip/hip_runtime.h>
#include <hip/hip_bf16.h>

#define N_ATOMS 100000
#define E_EDGES 200000
#define N2_SETS 200000
#define A_ASSN  400000
#define E2_EDGES 400000
#define G_MOL   5000
#define DIM     256
#define DIM2    512
#define FEAT    40
#define KEMB    64
#define NI2V    8
#define K2PAD   288
#define NLAYER  5
#define NBOND   4
#define BN_EPS  1e-5f

// epilogue flag bits
#define FB 1   // + bias
#define FA 2   // + addbuf
#define FR 4   // relu
#define FF 8   // write f32
#define FO 16  // write bf16

typedef __attribute__((ext_vector_type(16))) __bf16  v16bf;
typedef __attribute__((ext_vector_type(8)))  float   v8f;
typedef __attribute__((ext_vector_type(4)))  unsigned u32x4;

union Frag { v16bf v; u32x4 q[2]; };

__device__ __forceinline__ unsigned short f2bf(float f) {
  union { float f; unsigned u; } x; x.f = f;
  unsigned r = x.u + 0x7FFFu + ((x.u >> 16) & 1u);
  return (unsigned short)(r >> 16);
}

// ---------------------------------------------------------------------------
// WMMA bf16 GEMM:  C[M,Nc] = A[M,K](bf16,row-major) x BT[Nc,K](bf16, weights
// pre-transposed) with compile-time epilogue (FLAGS). K compile-time (mult of
// 32), Nc mult of 32. 8 waves/block; each wave owns a 16x32 output tile (two
// accumulators sharing one A fragment). Block tile 128x32. A row is CLAMPED
// (not predicated): rows >= M accumulate garbage that is never stored.
// ---------------------------------------------------------------------------
template <int K, int F>
__global__ void __launch_bounds__(256)
k_gemm_bf16(const unsigned short* __restrict__ A,
            const unsigned short* __restrict__ BT,
            const float* __restrict__ bias,
            const float* __restrict__ addbuf,
            float* __restrict__ outF,
            unsigned short* __restrict__ outB,
            int M, int Nc)
{
  constexpr bool HB = (F & FB) != 0;
  constexpr bool HA = (F & FA) != 0;
  constexpr bool HR = (F & FR) != 0;
  constexpr bool HF = (F & FF) != 0;
  constexpr bool HO = (F & FO) != 0;

  const int wave = threadIdx.x >> 5;
  const int lane = threadIdx.x & 31;
  const int tileRow = blockIdx.y * 128 + wave * 16;
  const int tileCol = blockIdx.x * 32;
  const int hi = lane >> 4;
  const int lo = lane & 15;

  int arow = tileRow + lo;
  arow = arow < M ? arow : (M - 1);                 // clamp, no divergence
  const unsigned short* aptr = A  + (size_t)arow * K;
  const unsigned short* b0   = BT + (size_t)(tileCol + lo) * K;
  const unsigned short* b1   = b0 + (size_t)16 * K;

  v8f acc0 = {0.f,0.f,0.f,0.f,0.f,0.f,0.f,0.f};
  v8f acc1 = {0.f,0.f,0.f,0.f,0.f,0.f,0.f,0.f};

#pragma unroll 4
  for (int k = 0; k < K; k += 32) {
    Frag a, p, q;
    // A 16x32 fragment: lane(hi,lo): K = hi*16 .. hi*16+15
    a.q[0] = *(const u32x4*)(aptr + k + hi * 16);
    a.q[1] = *(const u32x4*)(aptr + k + hi * 16 + 8);
    // B 32x16 fragments via BT: col=lo, K = hi*16 .. hi*16+15
    p.q[0] = *(const u32x4*)(b0 + k + hi * 16);
    p.q[1] = *(const u32x4*)(b0 + k + hi * 16 + 8);
    q.q[0] = *(const u32x4*)(b1 + k + hi * 16);
    q.q[1] = *(const u32x4*)(b1 + k + hi * 16 + 8);
    acc0 = __builtin_amdgcn_wmma_f32_16x16x32_bf16(
        false, a.v, false, p.v, (short)0, acc0, false, false);
    acc1 = __builtin_amdgcn_wmma_f32_16x16x32_bf16(
        false, a.v, false, q.v, (short)0, acc1, false, false);
  }

  // bias is invariant across the 8 accumulator rows: load once per lane
  float bias0 = 0.f, bias1 = 0.f;
  if (HB) { bias0 = bias[tileCol + lo]; bias1 = bias[tileCol + lo + 16]; }

  // C layout: VGPR r -> row = tileRow + hi*8 + r ; col = tileCol + lo (+16)
  const int row0 = tileRow + hi * 8;
  auto emit = [&](int r) {
    size_t idx = (size_t)(row0 + r) * Nc + tileCol + lo;
    float v0 = acc0[r] + bias0;
    float v1 = acc1[r] + bias1;
    if (HA) { v0 += addbuf[idx]; v1 += addbuf[idx + 16]; }
    if (HR) { v0 = v0 > 0.f ? v0 : 0.f; v1 = v1 > 0.f ? v1 : 0.f; }
    if (HF) { outF[idx] = v0; outF[idx + 16] = v1; }
    if (HO) { outB[idx] = f2bf(v0); outB[idx + 16] = f2bf(v1); }
  };

  if (tileRow + 16 <= M) {          // wave-uniform fast path: full tile
#pragma unroll
    for (int r = 0; r < 8; ++r) emit(r);
  } else {                          // tail row-block only
#pragma unroll
    for (int r = 0; r < 8; ++r)
      if (row0 + r < M) emit(r);
  }
}

// f32 [K,Nc] -> bf16 [Nc,Kpad] transposed, zero pad
__global__ void __launch_bounds__(256)
k_wprep(const float* __restrict__ W, unsigned short* __restrict__ WT,
        int K, int Nc, int Kpad)
{
  int t = blockIdx.x * 256 + threadIdx.x;
  if (t >= Nc * Kpad) return;
  int n = t / Kpad, k = t - n * Kpad;
  WT[t] = (k < K) ? f2bf(W[(size_t)k * Nc + n]) : (unsigned short)0;
}

// f32 [rows,cin] -> bf16 [rows,cout] row-major with zero pad (block = cout)
__global__ void k_cast_pad(const float* __restrict__ in,
                           unsigned short* __restrict__ out, int cin, int cout)
{
  int r = blockIdx.x, c = threadIdx.x;
  out[(size_t)r * cout + c] = (c < cin) ? f2bf(in[(size_t)r * cin + c])
                                        : (unsigned short)0;
}

// GIN message scatter: AGG[dst] += H[src] + table[attr]   (block = DIM)
__global__ void __launch_bounds__(DIM)
k_edge_scatter(const float* __restrict__ H, const int* __restrict__ ei,
               const int* __restrict__ attr, const float* __restrict__ tbl,
               float* __restrict__ AGG)
{
  int e = blockIdx.x, c = threadIdx.x;
  int s = ei[e], d = ei[E_EDGES + e], a = attr[e];
  float m = H[(size_t)s * DIM + c] + tbl[(size_t)a * DIM + c];
  atomicAdd(&AGG[(size_t)d * DIM + c], m);
}

// z0 = bf16((1+eps[l])*H + AGG)
__global__ void __launch_bounds__(256)
k_z0(const float* __restrict__ H, const float* __restrict__ AGG,
     const float* __restrict__ eps, int l, unsigned short* __restrict__ out,
     int total)
{
  int t = blockIdx.x * 256 + threadIdx.x;
  if (t >= total) return;
  float e = 1.0f + eps[l];
  out[t] = f2bf(e * H[t] + AGG[t]);
}

// column sums + sumsq for BN (block = DIM threads, chunk of 512 rows)
__global__ void __launch_bounds__(DIM)
k_colstats(const float* __restrict__ Z, float* __restrict__ stats, int M)
{
  int c = threadIdx.x;
  int r0 = blockIdx.x * 512;
  int r1 = r0 + 512 < M ? r0 + 512 : M;
  float s = 0.f, q = 0.f;
  for (int r = r0; r < r1; ++r) {
    float v = Z[(size_t)r * DIM + c];
    s += v; q += v * v;
  }
  atomicAdd(&stats[c], s);
  atomicAdd(&stats[DIM + c], q);
}

// BN apply (+relu) -> H f32
__global__ void __launch_bounds__(256)
k_bnapply(const float* __restrict__ Z, const float* __restrict__ stats,
          const float* __restrict__ gamma, const float* __restrict__ beta,
          float* __restrict__ H, int total, float invM, int relu)
{
  int t = blockIdx.x * 256 + threadIdx.x;
  if (t >= total) return;
  int c = t & (DIM - 1);
  float mu = stats[c] * invM;
  float var = stats[DIM + c] * invM - mu * mu;
  float v = gamma[c] * (Z[t] - mu) * rsqrtf(var + BN_EPS) + beta[c];
  if (relu) v = v > 0.f ? v : 0.f;
  H[t] = v;
}

// segment-sum scatter (block = DIM): sum[seg[r]] += src[gidx?gidx[r]:r]
__global__ void __launch_bounds__(DIM)
k_pool_scatter(const float* __restrict__ src, const int* __restrict__ seg,
               const int* __restrict__ gidx, float* __restrict__ sum,
               float* __restrict__ cnt)
{
  int r = blockIdx.x, c = threadIdx.x;
  int sr = gidx ? gidx[r] : r;
  int sg = seg[r];
  atomicAdd(&sum[(size_t)sg * DIM + c], src[(size_t)sr * DIM + c]);
  if (c == 0) atomicAdd(&cnt[sg], 1.0f);
}

// build X2B bf16 [N2, K2PAD] = [xp/cnt | iso | 0-pad]
__global__ void __launch_bounds__(K2PAD)
k_x2build(const float* __restrict__ XP, const float* __restrict__ cnt,
          const float* __restrict__ iso, unsigned short* __restrict__ X2B)
{
  int i = blockIdx.x, c = threadIdx.x;
  float v = 0.f;
  if (c < DIM) {
    float n = cnt[i]; n = n > 1.f ? n : 1.f;
    v = XP[(size_t)i * DIM + c] / n;
  } else if (c < DIM + NI2V) {
    v = iso[(size_t)i * NI2V + (c - DIM)];
  }
  X2B[(size_t)i * K2PAD + c] = f2bf(v);
}

// conv neighbor scatter: NEI[d2] += P[s2]   (block = DIM)
__global__ void __launch_bounds__(DIM)
k_nei_scatter(const float* __restrict__ P, const int* __restrict__ e2,
              float* __restrict__ NEI)
{
  int e = blockIdx.x, c = threadIdx.x;
  int s = e2[e], d = e2[E2_EDGES + e];
  atomicAdd(&NEI[(size_t)d * DIM + c], P[(size_t)s * DIM + c]);
}

// mcat bf16 [G,512] = [x1sum/c1 | x2sum/c2]
__global__ void __launch_bounds__(DIM2)
k_mcat(const float* __restrict__ x1s, const float* __restrict__ c1,
       const float* __restrict__ x2s, const float* __restrict__ c2,
       unsigned short* __restrict__ out)
{
  int g = blockIdx.x, c = threadIdx.x;
  float v;
  if (c < DIM) {
    float n = c1[g]; n = n > 1.f ? n : 1.f;
    v = x1s[(size_t)g * DIM + c] / n;
  } else {
    float n = c2[g]; n = n > 1.f ? n : 1.f;
    v = x2s[(size_t)g * DIM + (c - DIM)] / n;
  }
  out[(size_t)g * DIM2 + c] = f2bf(v);
}

// final: out[g] = m3[g,:] . last_w + last_b
__global__ void __launch_bounds__(256)
k_last(const float* __restrict__ m3, const float* __restrict__ w,
       const float* __restrict__ b, float* __restrict__ out, int G)
{
  int g = blockIdx.x * 256 + threadIdx.x;
  if (g >= G) return;
  float s = 0.f;
#pragma unroll
  for (int c = 0; c < 64; ++c) s += m3[(size_t)g * 64 + c] * w[c];
  out[g] = s + b[0];
}

// ---------------------------------------------------------------------------
template <int K, int F>
static inline void gemm(hipStream_t st, const unsigned short* A,
                        const unsigned short* BT, const float* bias,
                        const float* add, float* outF, unsigned short* outB,
                        int M, int Nc)
{
  dim3 g(Nc / 32, (M + 127) / 128);
  k_gemm_bf16<K, F><<<g, 256, 0, st>>>(A, BT, bias, add, outF, outB, M, Nc);
}

extern "C" void kernel_launch(void* const* d_in, const int* in_sizes, int n_in,
                              void* d_out, int out_size, void* d_ws, size_t ws_size,
                              hipStream_t stream)
{
  const float* x        = (const float*)d_in[0];
  const int*   ei       = (const int*)d_in[1];
  const int*   eattr    = (const int*)d_in[2];
  const float* iso2     = (const float*)d_in[3];
  const int*   ei2      = (const int*)d_in[4];
  const int*   ai2      = (const int*)d_in[5];
  const int*   batch    = (const int*)d_in[6];
  const int*   batch2   = (const int*)d_in[7];
  const float* W_emb    = (const float*)d_in[8];
  const float* b_emb    = (const float*)d_in[9];
  const float* etab     = (const float*)d_in[10];
  const float* eps      = (const float*)d_in[11];
  const float* gin_w1   = (const float*)d_in[12];
  const float* gin_b1   = (const float*)d_in[13];
  const float* gin_w2   = (const float*)d_in[14];
  const float* gin_b2   = (const float*)d_in[15];
  const float* bn_g     = (const float*)d_in[16];
  const float* bn_b     = (const float*)d_in[17];
  const float* iso1_ws  = (const float*)d_in[18];
  const float* iso1_wn  = (const float*)d_in[19];
  const float* iso1_b   = (const float*)d_in[20];
  const float* iso2_ws  = (const float*)d_in[21];
  const float* iso2_wn  = (const float*)d_in[22];
  const float* iso2_b   = (const float*)d_in[23];
  const float* fc0_w    = (const float*)d_in[24];
  const float* fc0_b    = (const float*)d_in[25];
  const float* fc1_w    = (const float*)d_in[26];
  const float* fc1_b    = (const float*)d_in[27];
  const float* fc2_w    = (const float*)d_in[28];
  const float* fc2_b    = (const float*)d_in[29];
  const float* last_w   = (const float*)d_in[30];
  const float* last_b   = (const float*)d_in[31];
  float* out = (float*)d_out;

  // ---- workspace arena --------------------------------------------------
  char* base = (char*)d_ws;
  size_t off = 0;
  auto carve = [&](size_t bytes) { char* p = base + off;
                                   off += (bytes + 255) & ~(size_t)255; return p; };
  float* H     = (float*)carve((size_t)N_ATOMS * DIM * 4);        // node feats
  float* AX    = (float*)carve((size_t)N2_SETS * DIM * 4);        // AGG / XP / x2 final
  float* ZP    = (float*)carve((size_t)N2_SETS * DIM * 4);        // GIN z / conv P
  unsigned short* BB1 = (unsigned short*)carve((size_t)N2_SETS * K2PAD * 2); // z1b / X2B
  unsigned short* BB0 = (unsigned short*)carve((size_t)N2_SETS * DIM * 2);   // xpad/z0b/X2B2
  float* NEI   = (float*)carve((size_t)N2_SETS * DIM * 4);
  float* stats = (float*)carve(2 * DIM * 4);
  float* x1sum = (float*)carve((size_t)G_MOL * DIM * 4);
  float* x2sum = (float*)carve((size_t)G_MOL * DIM * 4);
  float* cg1   = (float*)carve(G_MOL * 4);
  float* cg2   = (float*)carve(G_MOL * 4);
  float* cn2   = (float*)carve(N2_SETS * 4);
  unsigned short* mcat = (unsigned short*)carve((size_t)G_MOL * DIM2 * 2);
  unsigned short* m1b  = (unsigned short*)carve((size_t)G_MOL * DIM * 2);
  unsigned short* m2b  = (unsigned short*)carve((size_t)G_MOL * 128 * 2);
  float* m3f = (float*)carve((size_t)G_MOL * 64 * 4);
  unsigned short* wembT = (unsigned short*)carve(DIM * KEMB * 2);
  unsigned short* w1T   = (unsigned short*)carve(DIM2 * DIM * 2);
  unsigned short* w2T   = (unsigned short*)carve(DIM * DIM2 * 2);
  unsigned short* ws1T  = (unsigned short*)carve(DIM * K2PAD * 2);
  unsigned short* wn1T  = (unsigned short*)carve(DIM * K2PAD * 2);
  unsigned short* ws2T  = (unsigned short*)carve(DIM * DIM * 2);
  unsigned short* wn2T  = (unsigned short*)carve(DIM * DIM * 2);
  unsigned short* f0T   = (unsigned short*)carve(DIM * DIM2 * 2);
  unsigned short* f1T   = (unsigned short*)carve(128 * DIM * 2);
  unsigned short* f2T   = (unsigned short*)carve(64 * 128 * 2);

  auto wprep = [&](const float* W, unsigned short* WT, int K, int Nc, int Kpad) {
    int tot = Nc * Kpad;
    k_wprep<<<(tot + 255) / 256, 256, 0, stream>>>(W, WT, K, Nc, Kpad);
  };

  // ---- static weight prep ----------------------------------------------
  wprep(W_emb,   wembT, FEAT, DIM, KEMB);
  wprep(iso1_ws, ws1T,  DIM + NI2V, DIM, K2PAD);
  wprep(iso1_wn, wn1T,  DIM + NI2V, DIM, K2PAD);
  wprep(iso2_ws, ws2T,  DIM, DIM, DIM);
  wprep(iso2_wn, wn2T,  DIM, DIM, DIM);
  wprep(fc0_w,   f0T,   DIM2, DIM, DIM2);
  wprep(fc1_w,   f1T,   DIM, 128, DIM);
  wprep(fc2_w,   f2T,   128, 64, 128);

  // ---- atom embedding: H = relu(x @ W_emb + b) --------------------------
  k_cast_pad<<<N_ATOMS, KEMB, 0, stream>>>(x, BB0, FEAT, KEMB);
  gemm<KEMB, FB|FR|FF>(stream, BB0, wembT, b_emb, nullptr, H, nullptr,
                       N_ATOMS, DIM);

  // ---- GIN stack --------------------------------------------------------
  for (int l = 0; l < NLAYER; ++l) {
    wprep(gin_w1 + (size_t)l * DIM * DIM2, w1T, DIM, DIM2, DIM);
    wprep(gin_w2 + (size_t)l * DIM2 * DIM, w2T, DIM2, DIM, DIM2);
    hipMemsetAsync(AX, 0, (size_t)N_ATOMS * DIM * 4, stream);
    k_edge_scatter<<<E_EDGES, DIM, 0, stream>>>(
        H, ei, eattr, etab + (size_t)l * NBOND * DIM, AX);
    k_z0<<<(N_ATOMS * DIM + 255) / 256, 256, 0, stream>>>(
        H, AX, eps, l, BB0, N_ATOMS * DIM);
    gemm<DIM, FB|FR|FO>(stream, BB0, w1T, gin_b1 + (size_t)l * DIM2, nullptr,
                        nullptr, BB1, N_ATOMS, DIM2);
    gemm<DIM2, FB|FF>(stream, BB1, w2T, gin_b2 + (size_t)l * DIM, nullptr,
                      ZP, nullptr, N_ATOMS, DIM);
    hipMemsetAsync(stats, 0, 2 * DIM * 4, stream);
    k_colstats<<<(N_ATOMS + 511) / 512, DIM, 0, stream>>>(ZP, stats, N_ATOMS);
    k_bnapply<<<(N_ATOMS * DIM + 255) / 256, 256, 0, stream>>>(
        ZP, stats, bn_g + (size_t)l * DIM, bn_b + (size_t)l * DIM,
        H, N_ATOMS * DIM, 1.0f / (float)N_ATOMS, l < NLAYER - 1 ? 1 : 0);
  }

  // ---- graph pooling x_1 ------------------------------------------------
  hipMemsetAsync(x1sum, 0, (size_t)G_MOL * DIM * 4, stream);
  hipMemsetAsync(cg1, 0, G_MOL * 4, stream);
  k_pool_scatter<<<N_ATOMS, DIM, 0, stream>>>(H, batch, nullptr, x1sum, cg1);

  // ---- avg_pool onto 2-sets ---------------------------------------------
  hipMemsetAsync(AX, 0, (size_t)N2_SETS * DIM * 4, stream);
  hipMemsetAsync(cn2, 0, N2_SETS * 4, stream);
  k_pool_scatter<<<A_ASSN, DIM, 0, stream>>>(H, ai2 + A_ASSN, ai2, AX, cn2);
  k_x2build<<<N2_SETS, K2PAD, 0, stream>>>(AX, cn2, iso2, BB1);

  // ---- GraphConv 1 ------------------------------------------------------
  gemm<K2PAD, FF>(stream, BB1, wn1T, nullptr, nullptr, ZP, nullptr,
                  N2_SETS, DIM);
  hipMemsetAsync(NEI, 0, (size_t)N2_SETS * DIM * 4, stream);
  k_nei_scatter<<<E2_EDGES, DIM, 0, stream>>>(ZP, ei2, NEI);
  gemm<K2PAD, FB|FA|FR|FO>(stream, BB1, ws1T, iso1_b, NEI, nullptr, BB0,
                           N2_SETS, DIM);

  // ---- GraphConv 2 ------------------------------------------------------
  gemm<DIM, FF>(stream, BB0, wn2T, nullptr, nullptr, ZP, nullptr,
                N2_SETS, DIM);
  hipMemsetAsync(NEI, 0, (size_t)N2_SETS * DIM * 4, stream);
  k_nei_scatter<<<E2_EDGES, DIM, 0, stream>>>(ZP, ei2, NEI);
  gemm<DIM, FB|FA|FR|FF>(stream, BB0, ws2T, iso2_b, NEI, AX, nullptr,
                         N2_SETS, DIM);

  // ---- graph pooling x_2 ------------------------------------------------
  hipMemsetAsync(x2sum, 0, (size_t)G_MOL * DIM * 4, stream);
  hipMemsetAsync(cg2, 0, G_MOL * 4, stream);
  k_pool_scatter<<<N2_SETS, DIM, 0, stream>>>(AX, batch2, nullptr, x2sum, cg2);

  // ---- readout MLP ------------------------------------------------------
  k_mcat<<<G_MOL, DIM2, 0, stream>>>(x1sum, cg1, x2sum, cg2, mcat);
  gemm<DIM2, FB|FR|FO>(stream, mcat, f0T, fc0_b, nullptr, nullptr, m1b,
                       G_MOL, DIM);
  gemm<DIM, FB|FR|FO>(stream, m1b, f1T, fc1_b, nullptr, nullptr, m2b,
                      G_MOL, 128);
  gemm<128, FB|FR|FF>(stream, m2b, f2T, fc2_b, nullptr, m3f, nullptr,
                      G_MOL, 64);
  k_last<<<(G_MOL + 255) / 256, 256, 0, stream>>>(m3f, last_w, last_b, out, G_MOL);
}
